// WindowAttention_31834297598009
// MI455X (gfx1250) — matrix operand order repaired and checked by reading, not compiled, not run on hardware
//
#include <hip/hip_runtime.h>
#include <hip/hip_bf16.h>

#define DIM   192
#define NH    6
#define NTOK  98
#define NPAD  112

typedef __attribute__((ext_vector_type(16))) __bf16 v16bf;
typedef __attribute__((ext_vector_type(8)))  float  v8f;

union BF16x16 {
  v16bf v;
  uint4 q[2];
};

// LDS element offsets (bf16 elements)
#define XQ_OFF   0                         // [112][192]: X, then Q (scaled)
#define K_OFF    (NPAD*DIM)                // [112][192]
#define VT_OFF   (2*NPAD*DIM)              // [6][32][112] V transposed per head
#define ATTN_OFF (2*NPAD*DIM + NH*32*NPAD) // [112][112] softmax probs (bf16)
#define OUTH_OFF (ATTN_OFF + NPAD*NPAD)    // [112][32]  per-head attention out
#define LDS_ELEMS (OUTH_OFF + NPAD*32)     // 80640 elems = 161280 bytes

#define WMMA_BF16(a, b, c) \
  __builtin_amdgcn_wmma_f32_16x16x32_bf16(false, (a), false, (b), (short)0, (c), false, false)

// A-frag (16x32 bf16): lane holds row M=l16; elems i at K = half*16 + i (16 contiguous)
__device__ __forceinline__ v16bf ld_afrag(const __bf16* p) {
  BF16x16 r;
  r.q[0] = *(const uint4*)(p);
  r.q[1] = *(const uint4*)(p + 8);
  return r.v;
}
// B-frag (32x16 bf16): lane holds col N=l16; elems i at K = half*16 + i (16 contiguous)
__device__ __forceinline__ v16bf ld_bfrag_lds(const __bf16* p) {
  BF16x16 r;
  r.q[0] = *(const uint4*)(p);
  r.q[1] = *(const uint4*)(p + 8);
  return r.v;
}
// B-frag from global fp32 weights: 16 contiguous floats -> bf16
__device__ __forceinline__ v16bf ld_bfrag_gf32(const float* p) {
  float4 f0 = *(const float4*)(p + 0);
  float4 f1 = *(const float4*)(p + 4);
  float4 f2 = *(const float4*)(p + 8);
  float4 f3 = *(const float4*)(p + 12);
  BF16x16 r;
  r.v[0]  = (__bf16)f0.x; r.v[1]  = (__bf16)f0.y; r.v[2]  = (__bf16)f0.z; r.v[3]  = (__bf16)f0.w;
  r.v[4]  = (__bf16)f1.x; r.v[5]  = (__bf16)f1.y; r.v[6]  = (__bf16)f1.z; r.v[7]  = (__bf16)f1.w;
  r.v[8]  = (__bf16)f2.x; r.v[9]  = (__bf16)f2.y; r.v[10] = (__bf16)f2.z; r.v[11] = (__bf16)f2.w;
  r.v[12] = (__bf16)f3.x; r.v[13] = (__bf16)f3.y; r.v[14] = (__bf16)f3.z; r.v[15] = (__bf16)f3.w;
  return r.v;
}

__global__ void __launch_bounds__(224)
win_attn_fused(const float* __restrict__ x,
               const float* __restrict__ qkv_w,   // [576][192]
               const float* __restrict__ qkv_b,   // [576]
               const float* __restrict__ proj_w,  // [192][192]
               const float* __restrict__ proj_b,  // [192]
               const float* __restrict__ bias_table, // [507][6]
               const int*   __restrict__ rel_idx,    // [98][98]
               float* __restrict__ out)           // [B][98][192]
{
  __shared__ __attribute__((aligned(16))) __bf16 lds[LDS_ELEMS];

  const int tid  = threadIdx.x;
  const int lane = tid & 31;
  const int m    = tid >> 5;      // wave id == M-tile band (0..6)
  const int half = lane >> 4;
  const int l16  = lane & 15;
  const int b    = blockIdx.x;

  // ---- Phase 0: zero pad rows of X, load x (f32 -> bf16) into LDS ----
  {
    unsigned int* z = (unsigned int*)(lds + XQ_OFF + NTOK * DIM);
    for (int i = tid; i < (NPAD - NTOK) * DIM / 2; i += 224) z[i] = 0u;
    const float4* xg = (const float4*)(x + (size_t)b * NTOK * DIM);
    for (int i = tid; i < NTOK * DIM / 4; i += 224) {
      float4 f = xg[i];
      int row = i / (DIM / 4);
      int c   = (i % (DIM / 4)) * 4;
      union { __bf16 h[4]; uint2 u; } t;
      t.h[0] = (__bf16)f.x; t.h[1] = (__bf16)f.y; t.h[2] = (__bf16)f.z; t.h[3] = (__bf16)f.w;
      *(uint2*)(lds + XQ_OFF + row * DIM + c) = t.u;
    }
  }
  __syncthreads();

  const int arow = m * 16 + l16;  // A-frag row this lane owns

  // ---- Phase 1: QKV = X @ qkv_w^T + qkv_b (B-frags streamed from global) ----
  v8f qreg[12];
#pragma unroll
  for (int n = 0; n < 12; ++n) {           // Q columns 0..191
    float bv = qkv_b[n * 16 + l16];
    v8f acc;
#pragma unroll
    for (int r = 0; r < 8; ++r) acc[r] = bv;
#pragma unroll
    for (int kc = 0; kc < 6; ++kc) {
      v16bf a  = ld_afrag(lds + XQ_OFF + arow * DIM + kc * 32 + half * 16);
      v16bf bb = ld_bfrag_gf32(qkv_w + (size_t)(n * 16 + l16) * DIM + kc * 32 + half * 16);
      acc = WMMA_BF16(a, bb, acc);
    }
    qreg[n] = acc;
  }
#pragma unroll
  for (int n = 12; n < 24; ++n) {          // K columns 192..383 -> K_lds
    float bv = qkv_b[n * 16 + l16];
    v8f acc;
#pragma unroll
    for (int r = 0; r < 8; ++r) acc[r] = bv;
#pragma unroll
    for (int kc = 0; kc < 6; ++kc) {
      v16bf a  = ld_afrag(lds + XQ_OFF + arow * DIM + kc * 32 + half * 16);
      v16bf bb = ld_bfrag_gf32(qkv_w + (size_t)(n * 16 + l16) * DIM + kc * 32 + half * 16);
      acc = WMMA_BF16(a, bb, acc);
    }
    int nn = n - 12;
#pragma unroll
    for (int r = 0; r < 8; ++r) {
      int row = m * 16 + r + half * 8;
      lds[K_OFF + row * DIM + nn * 16 + l16] = (__bf16)acc[r];
    }
  }
#pragma unroll
  for (int n = 24; n < 36; ++n) {          // V columns 384..575 -> Vt_lds (transposed)
    float bv = qkv_b[n * 16 + l16];
    v8f acc;
#pragma unroll
    for (int r = 0; r < 8; ++r) acc[r] = bv;
#pragma unroll
    for (int kc = 0; kc < 6; ++kc) {
      v16bf a  = ld_afrag(lds + XQ_OFF + arow * DIM + kc * 32 + half * 16);
      v16bf bb = ld_bfrag_gf32(qkv_w + (size_t)(n * 16 + l16) * DIM + kc * 32 + half * 16);
      acc = WMMA_BF16(a, bb, acc);
    }
    int nn   = n - 24;
    int head = nn >> 1;
    int hd   = (nn & 1) * 16 + l16;
#pragma unroll
    for (int r = 0; r < 8; ++r) {
      int tok = m * 16 + r + half * 8;
      lds[VT_OFF + head * (32 * NPAD) + hd * NPAD + tok] = (__bf16)acc[r];
    }
  }
  // Q (scaled) written in place over X: band-private, this wave's X reads are done
  const float qscale = 0.17677669529663687f;  // 1/sqrt(32)
#pragma unroll
  for (int n = 0; n < 12; ++n) {
#pragma unroll
    for (int r = 0; r < 8; ++r) {
      int row = m * 16 + r + half * 8;
      lds[XQ_OFF + row * DIM + n * 16 + l16] = (__bf16)(qreg[n][r] * qscale);
    }
  }
  __syncthreads();  // K/Vt written by all waves, read by all waves below

  // ---- Phase 2: per-head attention + fused proj accumulation ----
  v8f pacc[12];
#pragma unroll
  for (int nt = 0; nt < 12; ++nt)
#pragma unroll
    for (int r = 0; r < 8; ++r) pacc[nt][r] = 0.0f;

  for (int h = 0; h < NH; ++h) {
    // S = Q_h @ K_h^T  (K-dim = 32 -> one WMMA per N-tile)
    v16bf qa = ld_afrag(lds + XQ_OFF + arow * DIM + h * 32 + half * 16);
    v8f s[7];
#pragma unroll
    for (int t = 0; t < 7; ++t) {
      v16bf kb = ld_bfrag_lds(lds + K_OFF + (t * 16 + l16) * DIM + h * 32 + half * 16);
      v8f z;
#pragma unroll
      for (int r = 0; r < 8; ++r) z[r] = 0.0f;
      s[t] = WMMA_BF16(qa, kb, z);
    }
    // relative-position bias gather + padding mask
#pragma unroll
    for (int t = 0; t < 7; ++t) {
#pragma unroll
      for (int r = 0; r < 8; ++r) {
        int i = m * 16 + r + half * 8;
        int j = t * 16 + l16;
        float v = s[t][r];
        if (j < NTOK) {
          if (i < NTOK) v += bias_table[rel_idx[i * NTOK + j] * NH + h];
        } else {
          v = -1e30f;
        }
        s[t][r] = v;
      }
    }
    // row softmax (row spans 16 lanes of one half + 7 frags) -> P (bf16) into attn LDS
#pragma unroll
    for (int r = 0; r < 8; ++r) {
      float mx = -1e30f;
#pragma unroll
      for (int t = 0; t < 7; ++t) mx = fmaxf(mx, s[t][r]);
      mx = fmaxf(mx, __shfl_xor(mx, 1, 32));
      mx = fmaxf(mx, __shfl_xor(mx, 2, 32));
      mx = fmaxf(mx, __shfl_xor(mx, 4, 32));
      mx = fmaxf(mx, __shfl_xor(mx, 8, 32));
      float sm = 0.0f;
#pragma unroll
      for (int t = 0; t < 7; ++t) {
        float p = __expf(s[t][r] - mx);
        s[t][r] = p;
        sm += p;
      }
      sm += __shfl_xor(sm, 1, 32);
      sm += __shfl_xor(sm, 2, 32);
      sm += __shfl_xor(sm, 4, 32);
      sm += __shfl_xor(sm, 8, 32);
      float inv = 1.0f / sm;
      int row = m * 16 + r + half * 8;
#pragma unroll
      for (int t = 0; t < 7; ++t)
        lds[ATTN_OFF + row * NPAD + t * 16 + l16] = (__bf16)(s[t][r] * inv);
    }

    // OUT_h(band) = P(band) @ V_h   (K = 112 padded to 128 with register zeros)
    v8f oacc[2];
#pragma unroll
    for (int v = 0; v < 2; ++v)
#pragma unroll
      for (int r = 0; r < 8; ++r) oacc[v][r] = 0.0f;
#pragma unroll
    for (int kc = 0; kc < 4; ++kc) {
      BF16x16 pa;
      const __bf16* ap = lds + ATTN_OFF + arow * NPAD + kc * 32 + half * 16;
      uint4 zz; zz.x = zz.y = zz.z = zz.w = 0u;
      pa.q[0] = (kc < 3 || half == 0) ? *(const uint4*)ap : zz;
      pa.q[1] = (kc < 3 || half == 0) ? *(const uint4*)(ap + 8) : zz;  // K>=112 -> 0
#pragma unroll
      for (int v = 0; v < 2; ++v) {
        BF16x16 vb;
        const __bf16* vp = lds + VT_OFF + h * (32 * NPAD) + (v * 16 + l16) * NPAD
                           + kc * 32 + half * 16;
        if (kc < 3 || half == 0) {
          vb.q[0] = *(const uint4*)vp;
          vb.q[1] = *(const uint4*)(vp + 8);
        } else {
          vb.q[0] = zz; vb.q[1] = zz;                       // tokens >=112 -> 0
        }
        oacc[v] = WMMA_BF16(pa.v, vb.v, oacc[v]);
      }
    }
    // stage OUT_h band in LDS (wave-private; same-wave DS ops are in order)
#pragma unroll
    for (int v = 0; v < 2; ++v)
#pragma unroll
      for (int r = 0; r < 8; ++r) {
        int row = m * 16 + r + half * 8;
        lds[OUTH_OFF + row * 32 + v * 16 + l16] = (__bf16)oacc[v][r];
      }
    // fused proj: pacc += OUT_h @ proj_w[:, h*32 : h*32+32]^T
    v16bf oa = ld_afrag(lds + OUTH_OFF + arow * 32 + half * 16);
#pragma unroll
    for (int nt = 0; nt < 12; ++nt) {
      v16bf pb = ld_bfrag_gf32(proj_w + (size_t)(nt * 16 + l16) * DIM + h * 32 + half * 16);
      pacc[nt] = WMMA_BF16(oa, pb, pacc[nt]);
    }
  }

  // ---- Phase 3: + proj_b, store f32 ----
  float* outp = out + (size_t)b * NTOK * DIM;
#pragma unroll
  for (int nt = 0; nt < 12; ++nt) {
    float pbv = proj_b[nt * 16 + l16];
#pragma unroll
    for (int r = 0; r < 8; ++r) {
      int i = m * 16 + r + half * 8;
      if (i < NTOK) outp[i * DIM + nt * 16 + l16] = pacc[nt][r] + pbv;
    }
  }
}

extern "C" void kernel_launch(void* const* d_in, const int* in_sizes, int n_in,
                              void* d_out, int out_size, void* d_ws, size_t ws_size,
                              hipStream_t stream) {
  const float* x          = (const float*)d_in[0];
  const float* qkv_w      = (const float*)d_in[1];
  const float* qkv_b      = (const float*)d_in[2];
  const float* proj_w     = (const float*)d_in[3];
  const float* proj_b     = (const float*)d_in[4];
  const float* bias_table = (const float*)d_in[5];
  const int*   rel_idx    = (const int*)d_in[6];
  float* out = (float*)d_out;

  int nwin = in_sizes[0] / (NTOK * DIM);  // 2048
  win_attn_fused<<<dim3(nwin), dim3(224), 0, stream>>>(
      x, qkv_w, qkv_b, proj_w, proj_b, bias_table, rel_idx, out);
}
